// ConvCapsule_83313775607877
// MI455X (gfx1250) — matrix operand order repaired and checked by reading, not compiled, not run on hardware
//
#include <hip/hip_runtime.h>
#include <hip/hip_bf16.h>
#include <math.h>

// ---------------- problem constants ----------------
#define Bn   8
#define IC   8
#define IDm  16
#define OC   8
#define OD   16
#define Himg 32
#define Wimg 32
#define NIMG (Bn*IC)          // 64
#define CIN  (IDm*4)          // 64
#define COUT (OC*OD*4)        // 512
#define NCH  (OC*OD)          // 128 BN channels
#define KTOT (CIN*9)          // 576
#define NPIX (NIMG*Himg*Wimg) // 65536

typedef _Float16 v16h __attribute__((ext_vector_type(16)));
typedef _Float16 v8h  __attribute__((ext_vector_type(8)));
typedef float    v8f  __attribute__((ext_vector_type(8)));

// ---------------------------------------------------------------------------
// 1) Weight transform: w(128,16,4,3,3) -> Bmat[co4=512][k=576] f16
//    co4 = ocod*4 + r ;  k = (kh*3+kw)*64 + cin ; cin = id*4+g
//    tw = rot90(roll(w, r, axis=group), k=r, axes=(kh,kw))
// ---------------------------------------------------------------------------
__global__ __launch_bounds__(256) void wtrans_kernel(
    const float* __restrict__ w, _Float16* __restrict__ Bmat) {
  int idx = blockIdx.x * 256 + threadIdx.x;
  if (idx >= COUT * KTOT) return;
  int co4 = idx / KTOT;
  int k   = idx % KTOT;
  int ocod = co4 >> 2, r = co4 & 3;
  int kpos = k / CIN, cin = k % CIN;
  int kh = kpos / 3, kw = kpos % 3;
  int id = cin >> 2, g = cin & 3;
  int gsrc = (g - r) & 3;                 // roll by r
  int sh = kh, sw = kw;                   // apply inverse rot90 r times
  for (int t = 0; t < r; ++t) { int nh = sw, nw = 2 - sh; sh = nh; sw = nw; }
  float val = w[(((ocod * IDm + id) * 4 + gsrc) * 3 + sh) * 3 + sw];
  Bmat[(size_t)co4 * KTOT + k] = (_Float16)val;
}

// ---------------------------------------------------------------------------
// 2) Implicit-GEMM conv via v_wmma_f32_16x16x32_f16.
//    Output layout P[pix][co], pix = n*1024 + h*32 + w  (routing-friendly,
//    coalesced stores, keeps the 134MB tensor streaming through L2).
//    Block: 256 thr = 8 waves = mt(2: w halves) x nt(4: 16-ch groups).
// ---------------------------------------------------------------------------
__global__ __launch_bounds__(256) void conv_kernel(
    const float* __restrict__ x, const _Float16* __restrict__ Bmat,
    float* __restrict__ P) {
  const int cotile = blockIdx.x;   // 0..7  -> co base = cotile*64
  const int h      = blockIdx.y;   // 0..31
  const int n      = blockIdx.z;   // 0..63
  __shared__ __align__(16) _Float16 xs[3][34][CIN];   // [row][w+1][cin], 13KB

  // stage 3 padded input rows as f16 (cin fastest -> conflict-free LDS writes)
  for (int i = threadIdx.x; i < 3 * 34 * CIN; i += 256) {
    int r   = i / (34 * CIN);
    int rem = i % (34 * CIN);
    int wp  = rem / CIN;
    int cin = rem % CIN;
    int hh = h - 1 + r, ww = wp - 1;
    float v = 0.f;
    if (hh >= 0 && hh < Himg && ww >= 0 && ww < Wimg)
      v = x[((size_t)(n * CIN + cin) * Himg + hh) * Wimg + ww];
    xs[r][wp][cin] = (_Float16)v;
  }
  __syncthreads();

  const int wv   = threadIdx.x >> 5;
  const int lane = threadIdx.x & 31;
  const int mt = wv & 1;          // which 16-pixel half of the row
  const int nt = wv >> 1;         // which 16-channel group
  const int hi = lane >> 4;       // lane half (selects K sub-range)
  const int mrow = (lane & 15) + mt * 16;        // A-matrix row -> output w
  const int co   = cotile * 64 + nt * 16 + (lane & 15);
  const _Float16* brow = Bmat + (size_t)co * KTOT;

  v8f acc = {};
  #pragma unroll
  for (int c = 0; c < 18; ++c) {               // K chunks of 32
    int kh   = c / 6;
    int kw   = (c % 6) >> 1;
    int half = c & 1;                           // which 32-cin half
    // A fragment (16-bit A 16x32 layout): two contiguous 8-half groups
    const _Float16* ap = &xs[kh][mrow + kw][half * 32 + hi * 8];
    v8h alo = *(const v8h*)ap;
    v8h ahi = *(const v8h*)(ap + 16);
    v16h a = __builtin_shufflevector(alo, ahi, 0,1,2,3,4,5,6,7,8,9,10,11,12,13,14,15);
    // B fragment: two contiguous 8-half K groups (same layout as A)
    const _Float16* bp = brow + c * 32 + hi * 8;
    v8h blo = *(const v8h*)bp;
    v8h bhi = *(const v8h*)(bp + 16);
    v16h bf = __builtin_shufflevector(blo, bhi, 0,1,2,3,4,5,6,7,8,9,10,11,12,13,14,15);
    acc = __builtin_amdgcn_wmma_f32_16x16x32_f16(
        false, a, false, bf, (short)0, acc, false, false);
  }

  // store C: lane col -> co, VGPR row v -> output w
  const size_t base = ((size_t)n * 1024 + (size_t)h * 32) * COUT;
  #pragma unroll
  for (int v = 0; v < 8; ++v) {
    int wout = v + hi * 8 + mt * 16;
    P[base + (size_t)wout * COUT + co] = acc[v];
  }
}

// ---------------------------------------------------------------------------
// 3) BN stats: one block per capsule channel (128), stats over N*4*H*W.
// ---------------------------------------------------------------------------
__global__ __launch_bounds__(256) void bnstats_kernel(
    const float* __restrict__ P, float* __restrict__ mean, float* __restrict__ rstd) {
  const int ch = blockIdx.x;        // 0..127
  const int t  = threadIdx.x;
  float s = 0.f, s2 = 0.f;
  for (int i = t; i < NPIX; i += 256) {
    size_t base = (size_t)i * COUT + ch * 4;
    #pragma unroll
    for (int g = 0; g < 4; ++g) { float v = P[base + g]; s += v; s2 += v * v; }
  }
  __shared__ float rs[256], rs2[256];
  rs[t] = s; rs2[t] = s2; __syncthreads();
  for (int o = 128; o > 0; o >>= 1) {
    if (t < o) { rs[t] += rs[t + o]; rs2[t] += rs2[t + o]; }
    __syncthreads();
  }
  if (t == 0) {
    const float inv = 1.f / (float)(NPIX * 4);
    float m = rs[0] * inv;
    float var = rs2[0] * inv - m * m;
    mean[ch] = m;
    rstd[ch] = rsqrtf(var + 1e-5f);
  }
}

// ---------------------------------------------------------------------------
// 4) Fused BN + dynamic routing + squash + entropy.
//    One block per (b,h,w) site, all 4 group elements together.
//    Thread t -> (g,ic,oc) = (t>>6, (t>>3)&7, t&7).
// ---------------------------------------------------------------------------
__global__ __launch_bounds__(256) void routing_kernel(
    const float* __restrict__ P, const float* __restrict__ mean,
    const float* __restrict__ rstd, const float* __restrict__ gamma,
    const float* __restrict__ beta, float* __restrict__ out) {
  const int w = blockIdx.x, h = blockIdx.y, b = blockIdx.z;
  const int t = threadIdx.x;
  __shared__ float pred[IC][COUT];          // [ic][(oc*16+od)*4+g]  16KB
  __shared__ float bij[4][IC][OC];
  __shared__ float cij[4][IC][OC];
  __shared__ float sj[4][OC][OD];
  __shared__ float vj[4][OC][OD];
  __shared__ float scl[4][OC];
  __shared__ float entarr[32];

  // coalesced load of all 8 in-capsules' predictions + BN
  for (int i = t; i < IC * COUT; i += 256) {
    int ic = i >> 9, co = i & 511;
    int n = b * IC + ic;
    float v = P[((size_t)n * 1024 + (size_t)h * 32 + w) * COUT + co];
    int ch = co >> 2;
    v = (v - mean[ch]) * rstd[ch] * gamma[ch] + beta[ch];
    pred[ic][co] = v;
  }
  const int g = t >> 6, ic = (t >> 3) & 7, oc = t & 7;
  bij[g][ic][oc] = 0.f;
  __syncthreads();

  for (int it = 0; it < 3; ++it) {
    // softmax over oc (per (g,ic))
    float bmax = -1e30f;
    #pragma unroll
    for (int o = 0; o < OC; ++o) bmax = fmaxf(bmax, bij[g][ic][o]);
    float e = __expf(bij[g][ic][oc] - bmax);
    cij[g][ic][oc] = e;
    __syncthreads();
    float sum = 0.f;
    #pragma unroll
    for (int o = 0; o < OC; ++o) sum += cij[g][ic][o];
    float cc = e / sum;
    __syncthreads();
    cij[g][ic][oc] = cc;
    __syncthreads();

    // s_j[g][oc][od] = sum_ic c * pred   (512 entries, 2 per thread)
    for (int i = t; i < 4 * OC * OD; i += 256) {
      int g2 = i >> 7, oc2 = (i >> 4) & 7, od = i & 15;
      float acc = 0.f;
      #pragma unroll
      for (int ic2 = 0; ic2 < IC; ++ic2)
        acc += cij[g2][ic2][oc2] * pred[ic2][(oc2 * OD + od) * 4 + g2];
      sj[g2][oc2][od] = acc;
    }
    __syncthreads();

    // squash scale per (g,oc)
    if (t < 32) {
      int g2 = t >> 3, oc2 = t & 7;
      float n2 = 0.f;
      #pragma unroll
      for (int od = 0; od < OD; ++od) { float xv = sj[g2][oc2][od]; n2 += xv * xv; }
      float nrm = sqrtf(n2);
      scl[g2][oc2] = n2 / (1.f + n2) / (nrm + 1e-8f);
    }
    __syncthreads();
    for (int i = t; i < 4 * OC * OD; i += 256) {
      int g2 = i >> 7, oc2 = (i >> 4) & 7, od = i & 15;
      vj[g2][oc2][od] = scl[g2][oc2] * sj[g2][oc2][od];
    }
    __syncthreads();

    if (it < 2) {        // agreement update
      float acc = 0.f;
      #pragma unroll
      for (int od = 0; od < OD; ++od)
        acc += pred[ic][(oc * OD + od) * 4 + g] * vj[g][oc][od];
      bij[g][ic][oc] += acc;
      __syncthreads();
    }
  }

  // entropy of final c over out_caps
  if (t < 32) {
    int g2 = t >> 3, ic2 = t & 7;
    float s = 0.f;
    #pragma unroll
    for (int o = 0; o < OC; ++o) { float c = cij[g2][ic2][o]; s -= c * __logf(c); }
    entarr[t] = s;
  }
  __syncthreads();
  if (t == 0) {
    float s = 0.f;
    for (int i = 0; i < 32; ++i) s += entarr[i];
    // /ln(OC) then mean over IC*4*H*W per batch, summed over batch
    atomicAdd(out + (size_t)Bn * OC * OD * 4 * Himg * Wimg,
              s / logf(8.f) * (1.f / 32768.f));
  }

  // write v_j: out[b][oc][od][g][h][w]
  for (int i = t; i < 4 * OC * OD; i += 256) {
    int g2 = i >> 7, oc2 = (i >> 4) & 7, od = i & 15;
    out[(((((size_t)b * OC + oc2) * OD + od) * 4 + g2) * Himg + h) * Wimg + w] =
        vj[g2][oc2][od];
  }
}

// ---------------------------------------------------------------------------
extern "C" void kernel_launch(void* const* d_in, const int* in_sizes, int n_in,
                              void* d_out, int out_size, void* d_ws, size_t ws_size,
                              hipStream_t stream) {
  (void)in_sizes; (void)n_in; (void)out_size; (void)ws_size;
  const float* x     = (const float*)d_in[0];   // (8,8,16,4,32,32)
  const float* w     = (const float*)d_in[1];   // (128,16,4,3,3)
  const float* gamma = (const float*)d_in[2];   // (128)
  const float* beta  = (const float*)d_in[3];   // (128)
  float* out = (float*)d_out;                   // 4194304 v_j + 1 entropy

  // workspace layout
  float*    P    = (float*)d_ws;                                   // 134217728 B
  _Float16* Bmat = (_Float16*)((char*)d_ws + (size_t)NPIX * COUT * 4);   // 589824 B
  float*    mean = (float*)((char*)Bmat + (size_t)COUT * KTOT * 2);
  float*    rstd = mean + NCH;

  // 1) weight transform
  wtrans_kernel<<<(COUT * KTOT + 255) / 256, 256, 0, stream>>>(w, Bmat);
  // 2) conv (WMMA implicit GEMM)
  conv_kernel<<<dim3(COUT / 64, Himg, NIMG), 256, 0, stream>>>(x, Bmat, P);
  // 3) BN stats
  bnstats_kernel<<<NCH, 256, 0, stream>>>(P, mean, rstd);
  // 4) zero the entropy accumulator slot
  hipMemsetAsync(out + (size_t)Bn * OC * OD * 4 * Himg * Wimg, 0, sizeof(float), stream);
  // 5) fused BN + routing + squash + entropy + output
  routing_kernel<<<dim3(Wimg, Himg, Bn), 256, 0, stream>>>(P, mean, rstd, gamma, beta, out);
}
